// Encoder_29454885716713
// MI455X (gfx1250) — matrix operand order repaired and checked by reading, not compiled, not run on hardware
//
#include <hip/hip_runtime.h>
#include <hip/hip_bf16.h>

// ---------------------------------------------------------------------------
// Fused column-softmax attention for MI455X (gfx1250, wave32, WMMA f16->f32)
//   q = (x Wq) * log2(e), k = x Wk, v = x Wv      (pass 0, VALU - tiny)
//   m'[k],Z[k] = colmax/colsum exp2(q' k^T)       (pass 1, WMMA streaming)
//   out = (exp2(q'k^T - m') @ (v/Z)) @ Wh         (pass 2, WMMA + LDS transpose)
// log2-domain softmax + raw v_exp_f32 (__builtin_amdgcn_exp2f): one TRANS op
// per exp, no log2e pre-multiply, no OCML denorm guard (inputs are <= 0 and
// sub-2^-126 softmax terms are negligible by construction).
// Score matrix (536 MB) is never materialized; per-batch q/k (128 KB f16)
// stays resident in the 192 MB L2 across the streaming passes.
// ---------------------------------------------------------------------------

typedef __attribute__((ext_vector_type(8)))  _Float16 v8h;
typedef __attribute__((ext_vector_type(16))) _Float16 v16h;
typedef __attribute__((ext_vector_type(8)))  float    v8f;

#define BATCH 32
#define SDIM  2048
#define FDIM  64
#define LOG2E 1.44269504088896340736f

static __device__ __forceinline__ float fast_exp2(float x) {
    return __builtin_amdgcn_exp2f(x);     // bare v_exp_f32 (flush-to-zero tail OK)
}
static __device__ __forceinline__ v8h ld8(const _Float16* p) {
    return *(const v8h*)p;   // 16B -> ds_load_b128 / global_load_b128
}
static __device__ __forceinline__ v16h make16(v8h lo, v8h hi) {
    return __builtin_shufflevector(lo, hi, 0,1,2,3,4,5,6,7,8,9,10,11,12,13,14,15);
}
static __device__ __forceinline__ v8f wmma_f16(v16h a, v16h b, v8f c) {
    // v_wmma_f32_16x16x32_f16  D = A(16x32 f16) * B(32x16 f16) + C(16x16 f32)
    return __builtin_amdgcn_wmma_f32_16x16x32_f16(false, a, false, b,
                                                  (short)0, c, false, false);
}

// ---------------------------------------------------------------------------
// Kernel 1: projections. One thread per sequence row.
// qh rows carry q*log2(e); qh/kh padded 24 -> 32 with zeros so one K=32 WMMA
// covers the whole d=24 contraction.
// ---------------------------------------------------------------------------
__global__ void k_proj(const float* __restrict__ x,
                       const float* __restrict__ Wq,
                       const float* __restrict__ Wk,
                       const float* __restrict__ Wv,
                       _Float16* __restrict__ qh,
                       _Float16* __restrict__ kh,
                       float*    __restrict__ vF)
{
    __shared__ float wS[64 * 80];           // [d][j]: j<24 Wq, 24..47 Wk, 48..79 Wv
    const int t = threadIdx.x;
    for (int i = t; i < 64 * 80; i += 256) {
        int d = i / 80, j = i % 80;
        float w;
        if (j < 24)      w = Wq[d * 24 + j];
        else if (j < 48) w = Wk[d * 24 + (j - 24)];
        else             w = Wv[d * 32 + (j - 48)];
        wS[i] = w;
    }
    __syncthreads();

    const int row = blockIdx.x * 256 + t;                 // < B*S (exact)
    const float* xr = x + (long)row * 64;

    float acc[80];
#pragma unroll
    for (int j = 0; j < 80; ++j) acc[j] = 0.f;
    for (int d = 0; d < 64; ++d) {
        const float xv = xr[d];
        const float* wd = &wS[d * 80];
#pragma unroll
        for (int j = 0; j < 80; ++j) acc[j] += xv * wd[j];
    }

    _Float16* qr = qh + (long)row * 32;
    _Float16* kr = kh + (long)row * 32;
    float*    vr = vF + (long)row * 32;
#pragma unroll
    for (int j = 0; j < 24; ++j) {
        qr[j] = (_Float16)(acc[j] * LOG2E);               // log2-domain scores
        kr[j] = (_Float16)acc[24 + j];
    }
#pragma unroll
    for (int j = 24; j < 32; ++j) { qr[j] = (_Float16)0.f; kr[j] = (_Float16)0.f; }
#pragma unroll
    for (int j = 0; j < 32; ++j) vr[j] = acc[48 + j];
}

// ---------------------------------------------------------------------------
// Kernel 2: per-key-column online softmax stats over the query axis.
// One wave owns a 16-column tile; streams 128 q-tiles, 1 WMMA each.
// Epilogue writes m'[k], Z[k] and v' = v/Z transposed to vt[b][d][S] (f16).
// ---------------------------------------------------------------------------
__global__ void k_colstats(const _Float16* __restrict__ qh,
                           const _Float16* __restrict__ kh,
                           const float*    __restrict__ vF,
                           float* __restrict__ mA,
                           float* __restrict__ zA,
                           _Float16* __restrict__ vt)
{
    const int b    = blockIdx.x;
    const int lane = threadIdx.x & 31;
    const int wave = threadIdx.x >> 5;
    const int col0 = (blockIdx.y * 8 + wave) * 16;
    const int half = lane >> 4, ln = lane & 15;

    // B-fragment of k^T for these 16 columns: lane n holds kh[col0+n][half*16 .. +16)
    const _Float16* kb = kh + ((long)(b * SDIM + col0 + ln) * 32 + half * 16);
    const v16h Bk = make16(ld8(kb), ld8(kb + 8));

    float mr = -1e30f, zr = 0.f;
    const _Float16* qbase = qh + ((long)(b * SDIM + ln) * 32 + half * 16);

    for (int qt = 0; qt < SDIM; qt += 16) {
        const _Float16* qp = qbase + (long)qt * 32;
        const v16h Aq = make16(ld8(qp), ld8(qp + 8));
        __builtin_prefetch(qp + 16 * 32, 0, 3);           // next q-tile
        v8f c = {};
        c = wmma_f16(Aq, Bk, c);                          // s'[m, col0+n] (log2 domain)

        // online column max / sum: column n lives in lanes n and n+16
        float tmax = c[0];
#pragma unroll
        for (int i = 1; i < 8; ++i) tmax = fmaxf(tmax, c[i]);
        tmax = fmaxf(tmax, __shfl_xor(tmax, 16, 32));
        const float mn = fmaxf(mr, tmax);
        float ps = 0.f;
#pragma unroll
        for (int i = 0; i < 8; ++i) ps += fast_exp2(c[i] - mn);   // bare v_exp_f32
        ps += __shfl_xor(ps, 16, 32);
        zr = zr * fast_exp2(mr - mn) + ps;
        mr = mn;
    }

    if (lane < 16) {
        const int k = col0 + ln;
        mA[b * SDIM + k] = mr;
        zA[b * SDIM + k] = zr;
        const float invZ = 1.0f / zr;
        const float* vrow = vF + (long)(b * SDIM + k) * 32;
#pragma unroll
        for (int d = 0; d < 32; ++d)
            vt[((long)b * 32 + d) * SDIM + k] = (_Float16)(vrow[d] * invZ);
    }
}

// ---------------------------------------------------------------------------
// Kernel 3: h1 = exp2(s' - m') @ v'  (k in chunks of 32 = two 16-col s tiles),
// then out = h1 @ Wh fused in the epilogue. One wave per 16-row q-tile.
// C->A transpose of P goes through per-wave LDS + explicit s_wait_dscnt
// (in-wave DS ordering makes a workgroup barrier unnecessary).
// ---------------------------------------------------------------------------
__global__ void k_attn_out(const _Float16* __restrict__ qh,
                           const _Float16* __restrict__ kh,
                           const _Float16* __restrict__ vt,
                           const float*    __restrict__ mA,
                           const float*    __restrict__ Wh,
                           float* __restrict__ out)
{
    __shared__ float whS[32 * 64];                        // 8 KB
    __shared__ __align__(16) _Float16 pS[8][16 * 32];     // 8 KB, per-wave P tile
    __shared__ __align__(16) float    hS[8][16 * 32];     // 16 KB, per-wave h1 tile

    const int t = threadIdx.x;
    for (int i = t; i < 32 * 64; i += 256) whS[i] = Wh[i];
    __syncthreads();

    const int b    = blockIdx.x;
    const int lane = t & 31;
    const int wave = t >> 5;
    const int q0   = (blockIdx.y * 8 + wave) * 16;
    const int half = lane >> 4, ln = lane & 15;

    const _Float16* qp = qh + ((long)(b * SDIM + q0 + ln) * 32 + half * 16);
    const v16h Aq = make16(ld8(qp), ld8(qp + 8));

    v8f h0 = {}, h1 = {};                                 // h1 cols d 0..15 / 16..31
    _Float16* pb = &pS[wave][0];
    const float* mrow = mA + b * SDIM;
    const int mb = half * 8;

    for (int k0 = 0; k0 < SDIM; k0 += 32) {
        // --- scores for columns k0..k0+31 (two WMMAs) ---
        const _Float16* kb0 = kh + ((long)(b * SDIM + k0 + ln) * 32 + half * 16);
        const _Float16* kb1 = kb0 + 16 * 32;
        const v16h Bk0 = make16(ld8(kb0), ld8(kb0 + 8));
        const v16h Bk1 = make16(ld8(kb1), ld8(kb1 + 8));
        v8f z = {};
        const v8f s0 = wmma_f16(Aq, Bk0, z);
        const v8f s1 = wmma_f16(Aq, Bk1, z);

        const float m0 = mrow[k0 + ln];
        const float m1 = mrow[k0 + 16 + ln];

        // --- P = exp2(s' - m') -> LDS row-major [16][32] f16 ---
#pragma unroll
        for (int i = 0; i < 8; ++i) {
            pb[(mb + i) * 32 + ln]      = (_Float16)fast_exp2(s0[i] - m0);
            pb[(mb + i) * 32 + 16 + ln] = (_Float16)fast_exp2(s1[i] - m1);
        }
        asm volatile("s_wait_dscnt 0" ::: "memory");      // in-wave DS order => P visible

        // --- reload P as an A-fragment (transposed layout) ---
        const _Float16* pr = pb + (long)ln * 32 + half * 16;
        const v16h Ap = make16(*(const v8h*)pr, *(const v8h*)(pr + 8));

        // --- v' B-fragments from transposed vt[b][d][S] ---
        const _Float16* vb0 = vt + ((long)(b * 32 + ln) * SDIM + k0 + half * 16);
        const _Float16* vb1 = vb0 + 16 * SDIM;
        const v16h Bv0 = make16(ld8(vb0), ld8(vb0 + 8));
        const v16h Bv1 = make16(ld8(vb1), ld8(vb1 + 8));

        h0 = wmma_f16(Ap, Bv0, h0);
        h1 = wmma_f16(Ap, Bv1, h1);
    }

    // --- epilogue: out[q0+r][:] = h1[r][:] @ Wh ---
    float* hb = &hS[wave][0];
#pragma unroll
    for (int i = 0; i < 8; ++i) {
        hb[(mb + i) * 32 + ln]      = h0[i];
        hb[(mb + i) * 32 + 16 + ln] = h1[i];
    }
    asm volatile("s_wait_dscnt 0" ::: "memory");

    float* orow = out + (long)(b * SDIM + q0) * 64;
    for (int tt = lane; tt < 16 * 64; tt += 32) {
        const int r = tt >> 6, f = tt & 63;
        float acc = 0.f;
#pragma unroll
        for (int d = 0; d < 32; ++d) acc += hb[r * 32 + d] * whS[d * 64 + f];
        orow[(long)r * 64 + f] = acc;
    }
}

// ---------------------------------------------------------------------------
extern "C" void kernel_launch(void* const* d_in, const int* in_sizes, int n_in,
                              void* d_out, int out_size, void* d_ws, size_t ws_size,
                              hipStream_t stream)
{
    const float* x  = (const float*)d_in[0];   // [32,2048,64]
    const float* Wq = (const float*)d_in[1];   // [64,24]
    const float* Wk = (const float*)d_in[2];   // [64,24]
    const float* Wv = (const float*)d_in[3];   // [64,32]
    const float* Wh = (const float*)d_in[4];   // [32,64]
    float* out = (float*)d_out;                // [32,2048,64]

    char* w = (char*)d_ws;
    const size_t NROW = (size_t)BATCH * SDIM;          // 65536
    _Float16* qh = (_Float16*)(w);                      // 4 MB  f16 [B,S,32]
    _Float16* kh = (_Float16*)(w + NROW * 32 * 2);      // 4 MB  f16 [B,S,32]
    _Float16* vt = (_Float16*)(w + NROW * 32 * 4);      // 4 MB  f16 [B,32,S]
    float*    vF = (float*)   (w + NROW * 32 * 6);      // 8 MB  f32 [B,S,32]
    float*    mA = (float*)   (w + NROW * 32 * 6 + NROW * 32 * 4);            // 256 KB
    float*    zA = (float*)   (w + NROW * 32 * 6 + NROW * 32 * 4 + NROW * 4); // 256 KB

    k_proj    <<<dim3(256),    dim3(256), 0, stream>>>(x, Wq, Wk, Wv, qh, kh, vF);
    k_colstats<<<dim3(32, 16), dim3(256), 0, stream>>>(qh, kh, vF, mA, zA, vt);
    k_attn_out<<<dim3(32, 16), dim3(256), 0, stream>>>(qh, kh, vt, mA, Wh, out);
}
